// SpiralAutoencoder_multiz_partkps_71734543778287
// MI455X (gfx1250) — compile-verified, matrix operand order checked by reading, not run on hardware
//
#include <hip/hip_runtime.h>
#include <cstdint>
#include <cstddef>

typedef __bf16 bf16;
typedef __attribute__((ext_vector_type(16))) __bf16 v16bf;
typedef __attribute__((ext_vector_type(8)))  __bf16 v8bf;
typedef __attribute__((ext_vector_type(8)))  float   v8f;

#define BATCH 32
#define N0 6891
#define N1 1724
#define N2 432
#define N3 108

// ---------------------------------------------------------------------------
// CDNA5 wave32 WMMA 16x16x32 bf16 fragment mapping (ISA 7.12.2):
//   A (16x32): lane L -> M=L%16, h=L/16;  K = (v%4)*2 + (v/4)*16 + h*8 + e
//   B (32x16): lane L -> N=L%16, same K packing (transpose-symmetric)
//   C/D (16x16 f32): lane L -> N=L%16; VGPR v -> M = v + 8*(L/16)
// Inverse (k -> in-lane position): h=(k>>3)&1, idx = ((k>>4)&1)*8+((k>>1)&3)*2+(k&1)
// => 8 consecutive, 8-aligned K values are CONTIGUOUS in one lane's fragment.
// Fragment storage unit: 32 lanes x 16 bf16 = 512 elems ("frag block").
// ---------------------------------------------------------------------------
__device__ __forceinline__ int k_to_half(int k) { return (k >> 3) & 1; }
__device__ __forceinline__ int k_to_idx(int k) {
  return ((k >> 4) & 1) * 8 + ((k >> 1) & 3) * 2 + (k & 1);
}

__device__ __forceinline__ v8bf zero_v8bf() {
  v8bf z;
#pragma unroll
  for (int i = 0; i < 8; ++i) z[i] = (bf16)0.0f;
  return z;
}

__device__ __forceinline__ v16bf combine16(v8bf lo, v8bf hi) {
  return __builtin_shufflevector(lo, hi, 0, 1, 2, 3, 4, 5, 6, 7,
                                 8, 9, 10, 11, 12, 13, 14, 15);
}

// two 16B vector loads -> one fragment
__device__ __forceinline__ v16bf load_frag16(const bf16* p) {
  const v8bf* q = (const v8bf*)p;
  return combine16(q[0], q[1]);
}

__device__ __forceinline__ float elu_f(float x) {
  return x > 0.0f ? x : (__expf(x) - 1.0f);
}

// ---------------------------------------------------------------------------
// Weight pre-pack kernels (run once per launch; stream-ordered, deterministic)
// ---------------------------------------------------------------------------
// A-operand pack: W (MxK f32 row-major) -> frag blocks [mt][kt][lane][16], zero-padded
__global__ void pack_a_frag_kernel(const float* __restrict__ W, bf16* __restrict__ out,
                                   int M, int K, int nmt, int nkt) {
  int idx = blockIdx.x * blockDim.x + threadIdx.x;
  if (idx >= nmt * nkt * 512) return;
  const int id = idx & 15;
  const int lane = (idx >> 4) & 31;
  const int kt = (idx >> 9) % nkt;
  const int mt = (idx >> 9) / nkt;
  const int m = mt * 16 + (lane & 15);
  const int h = lane >> 4;
  const int e = id & 1, v = id >> 1;
  const int k = kt * 32 + (v & 3) * 2 + (v >> 2) * 16 + h * 8 + e;
  bf16 val = (bf16)0.0f;
  if (m < M && k < K) val = (bf16)W[(size_t)m * K + k];
  out[idx] = val;
}

// B-operand pack (conv weights, transposed): W (OxKc f32) -> [kt][os][lane][16]
__global__ void pack_b_frag_kernel(const float* __restrict__ W, bf16* __restrict__ out,
                                   int O, int Kc, int NOS, int nkt) {
  int idx = blockIdx.x * blockDim.x + threadIdx.x;
  if (idx >= nkt * NOS * 512) return;
  const int id = idx & 15;
  const int lane = (idx >> 4) & 31;
  const int os = (idx >> 9) % NOS;
  const int kt = (idx >> 9) / NOS;
  const int o = os * 16 + (lane & 15);
  const int h = lane >> 4;
  const int e = id & 1, v = id >> 1;
  const int k = kt * 32 + (v & 3) * 2 + (v >> 2) * 16 + h * 8 + e;
  bf16 val = (bf16)0.0f;
  if (o < O && k < Kc) val = (bf16)W[(size_t)o * Kc + k];
  out[idx] = val;
}

// ---------------------------------------------------------------------------
// Dense GEMM (D/U einsums): C[m][c] = sum_k W[m][k]*H[k][c]
//   Ap : pre-packed A fragments of W       (direct global fragment loads)
//   H  : (K x NB) bf16 row-major activations, NB = 32*F (64B-aligned rows)
//   C  : (M x NB) bf16 row-major
// Block 256 = 8 waves; tile 128x64; wave w = M-subtile, 4 N-subtile accums.
// ---------------------------------------------------------------------------
__global__ __launch_bounds__(256)
void gemm_dense_frag_wmma(const bf16* __restrict__ Ap, const bf16* __restrict__ H,
                          bf16* __restrict__ C, int M, int K, int NB,
                          int nmt, int nkt) {
  __shared__ __align__(16) bf16 Bf[4 * 32 * 16];   // 4 N-subtiles of B fragments
  const int t = threadIdx.x, w = t >> 5, lane = t & 31;
  const int c0 = blockIdx.y * 64;
  const int mt = blockIdx.x * 8 + w;

  const v8f vzero = {0.f, 0.f, 0.f, 0.f, 0.f, 0.f, 0.f, 0.f};
  v8f acc[4] = {vzero, vzero, vzero, vzero};

  for (int kt = 0; kt < nkt; ++kt) {
    // stage B tile (32 x 64) into fragment-ordered LDS:
    // one global_load_b128 + 8 lane-scatter b16 stores per thread
    {
      const int k = t >> 3;              // 0..31
      const int nc0 = (t & 7) * 8;       // 0,8,...,56
      const int gk = kt * 32 + k;
      v8bf vals = zero_v8bf();
      if (gk < K) vals = *(const v8bf*)(H + (size_t)gk * NB + c0 + nc0);
      const int os = nc0 >> 4;
      const int nl0 = (nc0 & 15) + 16 * k_to_half(k);
      const int id = k_to_idx(k);
#pragma unroll
      for (int d = 0; d < 8; ++d)
        Bf[(os * 32 + nl0 + d) * 16 + id] = vals[d];
    }
    __syncthreads();

    v16bf afrag;
    if (mt < nmt) {
      const bf16* pa = Ap + ((size_t)mt * nkt + kt) * 512 + lane * 16;
      if (kt + 1 < nkt) __builtin_prefetch(pa + 512, 0, 1);
      afrag = load_frag16(pa);           // 2 x global_load_b128, coalesced
    } else {
      afrag = combine16(zero_v8bf(), zero_v8bf());
    }
#pragma unroll
    for (int s = 0; s < 4; ++s) {
      const v16bf bfrag = load_frag16(Bf + (s * 32 + lane) * 16);  // 2 x ds_load_b128
      acc[s] = __builtin_amdgcn_wmma_f32_16x16x32_bf16(
          false, afrag, false, bfrag, (short)0, acc[s], false, false);
    }
    __syncthreads();
  }

  const int hh = lane >> 4, nb = lane & 15;
#pragma unroll
  for (int s = 0; s < 4; ++s) {
#pragma unroll
    for (int v = 0; v < 8; ++v) {
      const int gm = blockIdx.x * 128 + w * 16 + v + 8 * hh;
      if (gm < M) C[(size_t)gm * NB + c0 + s * 16 + nb] = (bf16)acc[s][v];
    }
  }
}

// ---------------------------------------------------------------------------
// Fused spiral conv: out[(n,b)][o] = mask(n)*elu(sum_c Hin[S[n][j]][b][f]*W[o][c]+bias[o])
//   rows r = n*32+b (16-row subtile = 1 vertex x 16 batch lanes)
//   Wp: pre-packed B fragments of W^T (direct global loads, zero-padded)
//   A (gathered activations) staged in fragment-ordered LDS:
//     F%8==0 -> vector path: global_load_b128 + ds_store_b128 per 8-chunk
//     F==3 (enc0) -> scalar fallback
// ---------------------------------------------------------------------------
__global__ __launch_bounds__(256)
void spiral_conv_frag_wmma(const bf16* __restrict__ Hin, const int* __restrict__ S,
                           const bf16* __restrict__ Wp, const float* __restrict__ bias,
                           bf16* __restrict__ Hout, int N, int SP, int F, int O,
                           int NOS, int nkt) {
  __shared__ __align__(16) bf16 Af[8 * 32 * 16];   // 8 M-subtiles of A fragments
  const int Kc = F * SP;
  const int t = threadIdx.x, w = t >> 5, lane = t & 31;
  const int r0 = blockIdx.x * 128;
  const int o0 = blockIdx.y * 64;
  const bool vecpath = (F % 8) == 0;

  const v8f vzero = {0.f, 0.f, 0.f, 0.f, 0.f, 0.f, 0.f, 0.f};
  v8f acc[4] = {vzero, vzero, vzero, vzero};

  for (int kt = 0; kt < nkt; ++kt) {
    const int k0 = kt * 32;
    if (vecpath) {
      // 512 chunks of 8; 2 per thread: gather-load b128 -> fragment b128 store
#pragma unroll
      for (int cc = 0; cc < 2; ++cc) {
        const int cid = t * 2 + cc;
        const int r = cid >> 2;          // 0..127
        const int kc = (cid & 3) * 8;    // 0,8,16,24
        const int gr = r0 + r;
        const int n = gr >> 5, b = gr & 31;
        v8bf vals = zero_v8bf();
        if (n < N) {
          const int c = k0 + kc;         // < Kc (Kc multiple of 32 on this path)
          const int j = c / F;
          const int f0 = c - j * F;      // multiple of 8 -> 16B-aligned source
          const int src = S[n * SP + j];
          vals = *(const v8bf*)(Hin + ((size_t)src * BATCH + b) * F + f0);
        }
        const int ms = r >> 4, ml = r & 15;
        const int ln = ml + 16 * k_to_half(kc);
        *(v8bf*)(Af + (ms * 32 + ln) * 16 + ((kc >> 4) & 1) * 8) = vals;
      }
    } else {
      // scalar fallback (enc0: F=3, Kc=60)
      const int r = t >> 1;
      const int kk0 = (t & 1) * 16;
      const int gr = r0 + r;
      const int n = gr >> 5, b = gr & 31;
      const int ms = r >> 4, ml = r & 15;
#pragma unroll
      for (int i = 0; i < 16; ++i) {
        const int kk = kk0 + i;
        const int c = k0 + kk;
        bf16 val = (bf16)0.0f;
        if (n < N && c < Kc) {
          const int j = c / F, f = c - (c / F) * F;
          val = Hin[((size_t)S[n * SP + j] * BATCH + b) * F + f];
        }
        Af[(ms * 32 + ml + 16 * k_to_half(kk)) * 16 + k_to_idx(kk)] = val;
      }
    }
    __syncthreads();

    const v16bf afrag = load_frag16(Af + (w * 32 + lane) * 16);    // 2 x ds_load_b128
#pragma unroll
    for (int s = 0; s < 4; ++s) {
      const int os = (o0 >> 4) + s;
      const bf16* pb = Wp + ((size_t)kt * NOS + os) * 512 + lane * 16;
      if (kt + 1 < nkt) __builtin_prefetch(pb + (size_t)NOS * 512, 0, 1);
      const v16bf bfrag = load_frag16(pb);                         // 2 x global_load_b128
      acc[s] = __builtin_amdgcn_wmma_f32_16x16x32_bf16(
          false, afrag, false, bfrag, (short)0, acc[s], false, false);
    }
    __syncthreads();
  }

  const int hh = lane >> 4, nb = lane & 15;
#pragma unroll
  for (int s = 0; s < 4; ++s) {
#pragma unroll
    for (int v = 0; v < 8; ++v) {
      const int gr = r0 + w * 16 + v + 8 * hh;
      const int n = gr >> 5, b = gr & 31;
      const int o = o0 + s * 16 + nb;
      if (n < N && o < O) {
        float y = elu_f(acc[s][v] + bias[o]);
        if (n == N - 1) y = 0.0f;        // reference vertex mask
        Hout[((size_t)n * BATCH + b) * O + o] = (bf16)y;
      }
    }
  }
}

// ---------------------------------------------------------------------------
// Small tail kernels (bandwidth-trivial)
// ---------------------------------------------------------------------------
__global__ void pack_x_kernel(const float* __restrict__ x, bf16* __restrict__ H0) {
  int idx = blockIdx.x * blockDim.x + threadIdx.x;
  if (idx >= BATCH * N0 * 3) return;
  int f = idx % 3;
  int n = (idx / 3) % N0;
  int b = idx / (3 * N0);
  H0[((size_t)n * BATCH + b) * 3 + f] = (bf16)x[idx];
}

__global__ void fc_enc_kernel(const bf16* __restrict__ H3, const int* __restrict__ pidx,
                              const float* __restrict__ Wf, const float* __restrict__ bfv,
                              float* __restrict__ zout, int P, int k) {
  int idx = blockIdx.x * blockDim.x + threadIdx.x;
  if (idx >= BATCH * 64) return;
  int b = idx >> 6, o = idx & 63;
  const int Kf = P * 128;
  float acc = bfv[o];
  for (int p = 0; p < P; ++p) {
    const bf16* hrow = H3 + ((size_t)pidx[p] * BATCH + b) * 128;
    const float* wrow = Wf + (size_t)o * Kf + p * 128;
    for (int f = 0; f < 128; ++f) acc += (float)hrow[f] * wrow[f];
  }
  zout[b * (5 * 64) + k * 64 + o] = acc;
}

__global__ void kps_enc_kernel(const float* __restrict__ kps, const int* __restrict__ kidx,
                               const float* __restrict__ Wk, const float* __restrict__ bk,
                               float* __restrict__ zkout, int KS, int k) {
  int idx = blockIdx.x * blockDim.x + threadIdx.x;
  if (idx >= BATCH * 16) return;
  int b = idx >> 4, o = idx & 15;
  const int Kf = KS * 3;
  float acc = bk[o];
  for (int j = 0; j < KS; ++j) {
    const int kp = kidx[j];
    for (int c = 0; c < 3; ++c)
      acc += kps[((size_t)b * 24 + kp) * 3 + c] * Wk[(size_t)o * Kf + j * 3 + c];
  }
  zkout[b * (5 * 16) + k * 16 + o] = acc;
}

__global__ void fc_dec_kernel(const float* __restrict__ z, const float* __restrict__ zk,
                              const int* __restrict__ pidx,
                              const float* __restrict__ Wd, const float* __restrict__ bd,
                              bf16* __restrict__ H4, int P, int k) {
  int idx = blockIdx.x * blockDim.x + threadIdx.x;
  if (idx >= BATCH * P * 128) return;
  int f = idx % 128;
  int p = (idx / 128) % P;
  int b = idx / (128 * P);
  const int row = p * 128 + f;
  const float* wrow = Wd + (size_t)row * 80;
  const float* zin = z + b * 320 + k * 64;
  const float* zkin = zk + b * 80 + k * 16;
  float acc = bd[row];
  for (int i = 0; i < 64; ++i) acc += zin[i] * wrow[i];
  for (int i = 0; i < 16; ++i) acc += zkin[i] * wrow[64 + i];
  H4[((size_t)pidx[p] * BATCH + b) * 128 + f] = (bf16)acc;
}

__global__ void copy_dummy_kernel(const bf16* __restrict__ H3, bf16* __restrict__ H4) {
  int i = blockIdx.x * blockDim.x + threadIdx.x;
  if (i < BATCH * 128)
    H4[(size_t)(N3 - 1) * BATCH * 128 + i] = H3[(size_t)(N3 - 1) * BATCH * 128 + i];
}

__global__ void dec3_kernel(const bf16* __restrict__ Hin, const int* __restrict__ S0,
                            const float* __restrict__ W, const float* __restrict__ bias,
                            float* __restrict__ out) {
  int idx = blockIdx.x * blockDim.x + threadIdx.x;
  if (idx >= BATCH * N0 * 3) return;
  int c = idx % 3;
  int n = (idx / 3) % N0;
  float acc = 0.0f;
  if (n != N0 - 1) {
    int b = idx / (3 * N0);
    acc = bias[c];
    for (int j = 0; j < 20; ++j) {
      const int src = S0[n * 20 + j];
      const bf16* h = Hin + ((size_t)src * BATCH + b) * 32;
      const float* w = W + (size_t)c * 640 + j * 32;
      for (int f = 0; f < 32; ++f) acc += (float)h[f] * w[f];
    }
  }
  out[idx] = acc;  // idx == b*N0*3 + n*3 + c
}

// ---------------------------------------------------------------------------
// Host orchestration
// ---------------------------------------------------------------------------
static inline unsigned cdiv(long a, long b) { return (unsigned)((a + b - 1) / b); }

extern "C" void kernel_launch(void* const* d_in, const int* in_sizes, int n_in,
                              void* d_out, int out_size, void* d_ws, size_t ws_size,
                              hipStream_t stream) {
  (void)in_sizes; (void)n_in; (void)out_size; (void)ws_size;
  const float* x   = (const float*)d_in[0];
  const float* kps = (const float*)d_in[1];
  const float* Dm[3] = {(const float*)d_in[2], (const float*)d_in[3], (const float*)d_in[4]};
  const float* Um[3] = {(const float*)d_in[5], (const float*)d_in[6], (const float*)d_in[7]};
  const float* encW[3] = {(const float*)d_in[8],  (const float*)d_in[10], (const float*)d_in[12]};
  const float* encB[3] = {(const float*)d_in[9],  (const float*)d_in[11], (const float*)d_in[13]};
  const float* decW[4] = {(const float*)d_in[14], (const float*)d_in[16], (const float*)d_in[18], (const float*)d_in[20]};
  const float* decB[4] = {(const float*)d_in[15], (const float*)d_in[17], (const float*)d_in[19], (const float*)d_in[21]};
  const float *feW[5], *feB[5], *fdW[5], *fdB[5], *keW[5], *keB[5];
  for (int k = 0; k < 5; ++k) {
    feW[k] = (const float*)d_in[22 + 2 * k]; feB[k] = (const float*)d_in[23 + 2 * k];
    fdW[k] = (const float*)d_in[32 + 2 * k]; fdB[k] = (const float*)d_in[33 + 2 * k];
    keW[k] = (const float*)d_in[42 + 2 * k]; keB[k] = (const float*)d_in[43 + 2 * k];
  }
  const int* S0 = (const int*)d_in[52];
  const int* S1 = (const int*)d_in[53];
  const int* S2 = (const int*)d_in[54];
  const int* pidx[5] = {(const int*)d_in[55], (const int*)d_in[56], (const int*)d_in[57],
                        (const int*)d_in[58], (const int*)d_in[59]};
  const int* kidx[5] = {(const int*)d_in[60], (const int*)d_in[61], (const int*)d_in[62],
                        (const int*)d_in[63], (const int*)d_in[64]};
  float* out = (float*)d_out;
  float* z_out  = out + (size_t)BATCH * N0 * 3;      // (32,5,64)
  float* zk_out = z_out + (size_t)BATCH * 5 * 64;    // (32,5,16)

  // ---- workspace carve (256B aligned) ----
  char* base = (char*)d_ws;
  size_t off = 0;
  auto wsa = [&](size_t elems) -> bf16* {
    bf16* p = (bf16*)(base + off);
    off += (elems * sizeof(bf16) + 255) & ~(size_t)255;
    return p;
  };
  const int PSZ[5] = {25, 25, 25, 20, 12};
  const int KSZ[5] = {5, 5, 5, 5, 4};

  // dense (A-operand) packs: {M, K}
  const int dMK[6][2] = {{N1, N0}, {N2, N1}, {N3, N2},     // D0..D2
                         {N0, N1}, {N1, N2}, {N2, N3}};    // U0..U2
  const float* dSrc[6] = {Dm[0], Dm[1], Dm[2], Um[0], Um[1], Um[2]};
  bf16* dPack[6];
  int dNmt[6], dNkt[6];
  for (int i = 0; i < 6; ++i) {
    dNmt[i] = (dMK[i][0] + 15) / 16;
    dNkt[i] = (dMK[i][1] + 31) / 32;
    dPack[i] = wsa((size_t)dNmt[i] * dNkt[i] * 512);
  }
  // conv (B-operand) packs: {O, Kc}
  const int cOK[6][2] = {{32, 60}, {64, 448}, {128, 640},      // enc0..enc2
                         {64, 1280}, {32, 896}, {32, 640}};    // dec0..dec2
  const float* cSrc[6] = {encW[0], encW[1], encW[2], decW[0], decW[1], decW[2]};
  bf16* cPack[6];
  int cNOS[6], cNkt[6];
  for (int i = 0; i < 6; ++i) {
    cNOS[i] = 4 * cdiv(cOK[i][0], 64);
    cNkt[i] = (cOK[i][1] + 31) / 32;
    cPack[i] = wsa((size_t)cNkt[i] * cNOS[i] * 512);
  }

  bf16* H0 = wsa((size_t)N0 * BATCH * 3);
  bf16* A1 = wsa((size_t)N0 * BATCH * 32);
  bf16* H1 = wsa((size_t)N1 * BATCH * 32);
  bf16* A2 = wsa((size_t)N1 * BATCH * 64);
  bf16* H2 = wsa((size_t)N2 * BATCH * 64);
  bf16* A3 = wsa((size_t)N2 * BATCH * 128);
  bf16* H3 = wsa((size_t)N3 * BATCH * 128);
  bf16* H4 = wsa((size_t)N3 * BATCH * 128);
  bf16* B1 = wsa((size_t)N2 * BATCH * 128);
  bf16* B2 = wsa((size_t)N2 * BATCH * 64);
  bf16* B3 = wsa((size_t)N1 * BATCH * 64);
  bf16* B4 = wsa((size_t)N1 * BATCH * 32);
  bf16* B5 = wsa((size_t)N0 * BATCH * 32);
  bf16* B6 = wsa((size_t)N0 * BATCH * 32);

  // ---- weight packing ----
  for (int i = 0; i < 6; ++i) {
    long n = (long)dNmt[i] * dNkt[i] * 512;
    pack_a_frag_kernel<<<cdiv(n, 256), 256, 0, stream>>>(dSrc[i], dPack[i],
                                                         dMK[i][0], dMK[i][1], dNmt[i], dNkt[i]);
  }
  for (int i = 0; i < 6; ++i) {
    long n = (long)cNkt[i] * cNOS[i] * 512;
    pack_b_frag_kernel<<<cdiv(n, 256), 256, 0, stream>>>(cSrc[i], cPack[i],
                                                         cOK[i][0], cOK[i][1], cNOS[i], cNkt[i]);
  }

  auto dense = [&](int i, const bf16* H, bf16* C, int NB) {
    dim3 g(cdiv(dMK[i][0], 128), (unsigned)(NB / 64));
    gemm_dense_frag_wmma<<<g, 256, 0, stream>>>(dPack[i], H, C,
                                                dMK[i][0], dMK[i][1], NB, dNmt[i], dNkt[i]);
  };
  auto conv = [&](int i, const bf16* H, const int* S, const float* bias, bf16* Ho,
                  int N, int SP, int F) {
    const int O = cOK[i][0];
    dim3 g(cdiv((long)N * BATCH, 128), cdiv(O, 64));
    spiral_conv_frag_wmma<<<g, 256, 0, stream>>>(H, S, cPack[i], bias, Ho,
                                                 N, SP, F, O, cNOS[i], cNkt[i]);
  };

  // ---- encoder ----
  pack_x_kernel<<<cdiv((long)BATCH * N0 * 3, 256), 256, 0, stream>>>(x, H0);
  conv(0, H0, S0, encB[0], A1, N0, 20, 3);
  dense(0, A1, H1, BATCH * 32);
  conv(1, H1, S1, encB[1], A2, N1, 14, 32);
  dense(1, A2, H2, BATCH * 64);
  conv(2, H2, S2, encB[2], A3, N2, 10, 64);
  dense(2, A3, H3, BATCH * 128);

  // ---- latent heads ----
  for (int k = 0; k < 5; ++k)
    fc_enc_kernel<<<cdiv(BATCH * 64, 256), 256, 0, stream>>>(H3, pidx[k], feW[k], feB[k],
                                                             z_out, PSZ[k], k);
  for (int k = 0; k < 5; ++k)
    kps_enc_kernel<<<cdiv(BATCH * 16, 256), 256, 0, stream>>>(kps, kidx[k], keW[k], keB[k],
                                                              zk_out, KSZ[k], k);
  for (int k = 0; k < 5; ++k)
    fc_dec_kernel<<<cdiv((long)BATCH * PSZ[k] * 128, 256), 256, 0, stream>>>(
        z_out, zk_out, pidx[k], fdW[k], fdB[k], H4, PSZ[k], k);
  copy_dummy_kernel<<<cdiv(BATCH * 128, 256), 256, 0, stream>>>(H3, H4);

  // ---- decoder ----
  dense(5, H4, B1, BATCH * 128);           // U2
  conv(3, B1, S2, decB[0], B2, N2, 10, 128);
  dense(4, B2, B3, BATCH * 64);            // U1
  conv(4, B3, S1, decB[1], B4, N1, 14, 64);
  dense(3, B4, B5, BATCH * 32);            // U0
  conv(5, B5, S0, decB[2], B6, N0, 20, 32);
  dec3_kernel<<<cdiv((long)BATCH * N0 * 3, 256), 256, 0, stream>>>(B6, S0, decW[3], decB[3], out);
}
